// DecoderModel_54795192762653
// MI455X (gfx1250) — matrix operand order repaired and checked by reading, not compiled, not run on hardware
//
#include <hip/hip_runtime.h>
#include <hip/hip_bf16.h>

// ---------------------------------------------------------------------------
// Model dims (compile-time constants from the reference)
// ---------------------------------------------------------------------------
#define DD   1024          // model dim
#define HH   16            // query heads
#define KK   4             // kv heads
#define FF   4096          // ffn dim
#define LL   4             // layers
#define VV   32000         // vocab
#define TT   1024          // seq len
#define BB   4             // batch
#define HDM  64            // head dim
#define NQKV 1536          // (H + 2K) * HD

typedef _Float16 f16;
typedef __attribute__((ext_vector_type(16))) _Float16 v16h;
typedef __attribute__((ext_vector_type(8)))  _Float16 v8h;
typedef __attribute__((ext_vector_type(8)))  float    v8f;
typedef __attribute__((ext_vector_type(4)))  int      i32x4;

// ---- gfx1250 async global->LDS path (guarded: falls back if unsupported) ---
#if defined(__has_builtin)
#if __has_builtin(__builtin_amdgcn_global_load_async_to_lds_b128) && \
    __has_builtin(__builtin_amdgcn_s_wait_asynccnt)
#define ASYNC_LDS 1
#endif
#endif
#ifndef ASYNC_LDS
#define ASYNC_LDS 0
#endif

__device__ __forceinline__ void cp_glds16(f16* lds, const f16* g) {
#if ASYNC_LDS
  // GLOBAL_LOAD_ASYNC_TO_LDS_B128: LDS[lds] = MEM[g], tracked by ASYNCcnt.
  // Builtin signature (from compiler diagnostic): int32x4 ptrs, AS1 src / AS3 dst.
  __builtin_amdgcn_global_load_async_to_lds_b128(
      (__attribute__((address_space(1))) i32x4*)(const void*)g,
      (__attribute__((address_space(3))) i32x4*)(void*)lds,
      /*imm offset=*/0, /*cpol=*/0);
#else
  *(v8h*)lds = *(const v8h*)g;
#endif
}

__device__ __forceinline__ v8f wmma32(v16h a, v16h b, v8f c) {
  // D = A(16x32 f16) * B(32x16 f16) + C(16x16 f32)
  return __builtin_amdgcn_wmma_f32_16x16x32_f16(
      /*neg_a=*/false, a, /*neg_b=*/false, b,
      /*c_mod=*/(short)0, c, /*reuse_a=*/false, /*reuse_b=*/false);
}

// ---------------------------------------------------------------------------
// f32 -> f16 weight conversion
// ---------------------------------------------------------------------------
__global__ void cvt_kernel(const float* __restrict__ src, f16* __restrict__ dst,
                           size_t n) {
  size_t i = (size_t)blockIdx.x * blockDim.x + threadIdx.x;
  size_t stride = (size_t)gridDim.x * blockDim.x;
  for (; i < n; i += stride) dst[i] = (f16)src[i];
}

// ---------------------------------------------------------------------------
// Embedding * sqrt(D) + sinusoidal PE
// ---------------------------------------------------------------------------
__global__ void embed_kernel(const int* __restrict__ idx,
                             const float* __restrict__ emb,
                             float* __restrict__ h) {
  int row = blockIdx.x;            // b*T + t
  int t = row % TT;
  int token = idx[row];
  for (int d = threadIdx.x; d < DD; d += 256) {
    int i = d >> 1;
    float freq = __expf(-((float)(2 * i) / (float)DD) * 9.210340371976184f);
    float ang = (float)t * freq;
    float pe = (d & 1) ? __cosf(ang) : __sinf(ang);
    h[(size_t)row * DD + d] = emb[(size_t)token * DD + d] * 32.0f + pe;
  }
}

// ---------------------------------------------------------------------------
// LayerNorm: fp32 in, f16 out (feeds WMMA GEMMs)
// ---------------------------------------------------------------------------
__global__ __launch_bounds__(256) void layernorm_kernel(
    const float* __restrict__ h, const float* __restrict__ g,
    const float* __restrict__ bta, f16* __restrict__ xh) {
  __shared__ float r1[256];
  __shared__ float r2[256];
  int row = blockIdx.x;
  int tid = threadIdx.x;
  const float* hr = h + (size_t)row * DD;
  float s = 0.f, ss = 0.f;
  for (int i = tid; i < DD; i += 256) {
    float v = hr[i];
    s += v;
    ss += v * v;
  }
  r1[tid] = s; r2[tid] = ss;
  __syncthreads();
  for (int st = 128; st > 0; st >>= 1) {
    if (tid < st) { r1[tid] += r1[tid + st]; r2[tid] += r2[tid + st]; }
    __syncthreads();
  }
  float mean = r1[0] * (1.0f / DD);
  float var = r2[0] * (1.0f / DD) - mean * mean;
  float inv = rsqrtf(var + 1e-5f);
  for (int i = tid; i < DD; i += 256)
    xh[(size_t)row * DD + i] = (f16)(((hr[i] - mean) * inv) * g[i] + bta[i]);
}

// ---------------------------------------------------------------------------
// WMMA GEMM: C(f32) = A(f16, MxK) * B(f16, KxN) [+bias] [+resid], optional f16
// block = 256 threads = 8 waves; tile 128x128, BK=32; wave tile 32x64.
// Double-buffered LDS; async global->LDS copies when available, so tile t+1
// streams in (ASYNCcnt) while tile t feeds the v_wmma chain.
// ---------------------------------------------------------------------------
__global__ __launch_bounds__(256) void gemm_f16_kernel(
    const f16* __restrict__ A, const f16* __restrict__ Bw,
    int M, int N, int Kd,
    const float* __restrict__ bias, const float* __restrict__ resid,
    float* __restrict__ outF, f16* __restrict__ outH) {
  __shared__ f16 As[2][128 * 32];
  __shared__ f16 Bs[2][32 * 128];
  const int tid = threadIdx.x;
  const int lane = tid & 31;
  const int wid = tid >> 5;
  const int wm = (wid >> 1) * 32;   // 4 waves along M
  const int wn = (wid & 1) * 64;    // 2 waves along N
  const int bm = blockIdx.y * 128;
  const int bn = blockIdx.x * 128;

  // each thread stages 2x16B of A and 2x16B of B per tile (4 async copies)
  auto stage = [&](int k0, int buf) {
#pragma unroll
    for (int it = 0; it < 2; ++it) {
      int e = (tid + it * 256) * 8;
      int r = e >> 5, c = e & 31;
      cp_glds16(&As[buf][e], &A[(size_t)(bm + r) * Kd + k0 + c]);
    }
#pragma unroll
    for (int it = 0; it < 2; ++it) {
      int e = (tid + it * 256) * 8;
      int r = e >> 7, c = e & 127;
      cp_glds16(&Bs[buf][e], &Bw[(size_t)(k0 + r) * N + bn + c]);
    }
  };

  v8f acc[2][4] = {};
  const int nk = Kd >> 5;

  stage(0, 0);
  for (int t = 0; t < nk; ++t) {
    const int buf = t & 1;
    if (t + 1 < nk) {
      stage((t + 1) << 5, buf ^ 1);
#if ASYNC_LDS
      __builtin_amdgcn_s_wait_asynccnt(4);  // tile t done; t+1 still in flight
#endif
    } else {
#if ASYNC_LDS
      __builtin_amdgcn_s_wait_asynccnt(0);
#endif
    }
    __syncthreads();

    // ---- fragments: A and B both use contiguous k = 16*(lane>>4) + i ----
    v16h af[2], bf[4];
#pragma unroll
    for (int mi = 0; mi < 2; ++mi) {
      int row = wm + mi * 16 + (lane & 15);
      int kb = (lane >> 4) * 16;
#pragma unroll
      for (int i = 0; i < 16; ++i)
        af[mi][i] = As[buf][row * 32 + kb + i];
    }
#pragma unroll
    for (int ni = 0; ni < 4; ++ni) {
      int col = wn + ni * 16 + (lane & 15);
      int kb = (lane >> 4) * 16;
#pragma unroll
      for (int i = 0; i < 16; ++i)
        bf[ni][i] = Bs[buf][(kb + i) * 128 + col];
    }
#pragma unroll
    for (int mi = 0; mi < 2; ++mi)
#pragma unroll
      for (int ni = 0; ni < 4; ++ni)
        acc[mi][ni] = wmma32(af[mi], bf[ni], acc[mi][ni]);

    __syncthreads();  // all waves done with buf before it is re-staged
  }

  // ---- epilogue: C layout lane=(N), vgpr r -> row r+8*(lane>>4) ----
  const int half = lane >> 4;
  const int cn = lane & 15;
#pragma unroll
  for (int mi = 0; mi < 2; ++mi)
#pragma unroll
    for (int ni = 0; ni < 4; ++ni)
#pragma unroll
      for (int r = 0; r < 8; ++r) {
        int m = bm + wm + mi * 16 + r + 8 * half;
        int n = bn + wn + ni * 16 + cn;
        float v = acc[mi][ni][r];
        if (bias) v += bias[n];
        size_t o = (size_t)m * N + n;
        if (resid) v += resid[o];
        if (outF) outF[o] = v;
        if (outH) outH[o] = (f16)v;
      }
}

// ---------------------------------------------------------------------------
// Flash-style causal attention with WMMA (GQA with the reference's raw
// reshape index mapping):
//   q(b,g,kh,l,d) = qkv[b*T + head*64 + (l>>4)][(l&15)*64 + d],  head=g*K+kh
//   k(b,kh,j,d)   = qkv[b*T + kh*256  + (j>>2)][1024 + (j&3)*64 + d]
//   v(b,kh,j,d)   = qkv[b*T + kh*256  + (j>>2)][1280 + (j&3)*64 + d]
//   o row (b*T+l), col head*64+d
// grid = (T/128, B*H), block = 256 (8 waves, one 16-row q-tile per wave)
// ---------------------------------------------------------------------------
__global__ __launch_bounds__(256) void attention_kernel(
    const f16* __restrict__ qkv, f16* __restrict__ o) {
  __shared__ f16 Pl[8][16 * 32];   // per-wave P tile for C->A relayout
  const int tid = threadIdx.x, lane = tid & 31, wid = tid >> 5;
  const int bh = blockIdx.y;
  const int b = bh / HH, head = bh % HH;
  const int kh = head % KK;
  const int l0 = (blockIdx.x * 8 + wid) * 16;
  const float scale = 0.125f;  // 1/sqrt(64)
  const size_t rowBase = (size_t)b * TT;

  // ---- Q fragments (A layout): row = lane&15, d contiguous, 16*(lane>>4) + i ----
  v16h qf[2];
  {
    int lrow = l0 + (lane & 15);
    int trow = head * 64 + (lrow >> 4);
    size_t base = (rowBase + trow) * NQKV + (size_t)(lrow & 15) * 64;
#pragma unroll
    for (int kk = 0; kk < 2; ++kk) {
      int db = kk * 32 + (lane >> 4) * 16;
#pragma unroll
      for (int i = 0; i < 16; ++i)
        qf[kk][i] = qkv[base + db + i];
    }
  }

  float mx[8], ls[8];
#pragma unroll
  for (int r = 0; r < 8; ++r) { mx[r] = -1e30f; ls[r] = 0.0f; }
  v8f oac[4] = {};

  for (int jb = 0; jb < TT; jb += 32) {   // uniform trip count: safe barriers
    const bool active = (jb <= l0 + 15);
    if (active) {
      // ---- S = Q Kt for two 16-key column tiles ----
      v8f s0 = {}, s1 = {};
#pragma unroll
      for (int ct = 0; ct < 2; ++ct) {
        int j = jb + ct * 16 + (lane & 15);
        size_t kbase = (rowBase + kh * 256 + (j >> 2)) * NQKV + 1024 +
                       (size_t)(j & 3) * 64;
#pragma unroll
        for (int kk = 0; kk < 2; ++kk) {
          v16h kf;  // B layout: col=key(lane&15), k=d-slice
          int db = kk * 32 + (lane >> 4) * 16;
#pragma unroll
          for (int i = 0; i < 16; ++i) kf[i] = qkv[kbase + db + i];
          if (ct == 0) s0 = wmma32(qf[kk], kf, s0);
          else         s1 = wmma32(qf[kk], kf, s1);
        }
      }
      // ---- causal mask + online softmax ----
      int j0 = jb + (lane & 15), j1 = j0 + 16;
      float sm0[8], sm1[8], pm[8];
#pragma unroll
      for (int r = 0; r < 8; ++r) {
        int l = l0 + r + 8 * (lane >> 4);
        sm0[r] = (j0 <= l) ? s0[r] * scale : -1e30f;
        sm1[r] = (j1 <= l) ? s1[r] * scale : -1e30f;
        pm[r] = fmaxf(sm0[r], sm1[r]);
      }
#pragma unroll
      for (int off = 1; off < 16; off <<= 1)
#pragma unroll
        for (int r = 0; r < 8; ++r)
          pm[r] = fmaxf(pm[r], __shfl_xor(pm[r], off, 32));
      float p0[8], p1[8], rs[8];
#pragma unroll
      for (int r = 0; r < 8; ++r) {
        float mn = fmaxf(mx[r], pm[r]);
        float alpha = __expf(mx[r] - mn);
        mx[r] = mn;
        ls[r] *= alpha;
#pragma unroll
        for (int nt = 0; nt < 4; ++nt) oac[nt][r] *= alpha;
        p0[r] = __expf(sm0[r] - mn);
        p1[r] = __expf(sm1[r] - mn);
        rs[r] = p0[r] + p1[r];
      }
#pragma unroll
      for (int off = 1; off < 16; off <<= 1)
#pragma unroll
        for (int r = 0; r < 8; ++r)
          rs[r] += __shfl_xor(rs[r], off, 32);
#pragma unroll
      for (int r = 0; r < 8; ++r) ls[r] += rs[r];
      // ---- park P (C layout) into LDS ----
#pragma unroll
      for (int r = 0; r < 8; ++r) {
        int row = r + 8 * (lane >> 4);
        Pl[wid][row * 32 + (lane & 15)]      = (f16)p0[r];
        Pl[wid][row * 32 + 16 + (lane & 15)] = (f16)p1[r];
      }
    }
    __syncthreads();
    if (active) {
      // ---- reload P in A layout, O += P * V ----
      v16h pf;
      {
        int row = lane & 15;
        int kb = (lane >> 4) * 16;
#pragma unroll
        for (int i = 0; i < 16; ++i)
          pf[i] = Pl[wid][row * 32 + kb + i];
      }
#pragma unroll
      for (int nt = 0; nt < 4; ++nt) {
        v16h vf;  // B layout: col=d(lane&15), k=key
        int dcol = nt * 16 + (lane & 15);
        int keyb = jb + (lane >> 4) * 16;
#pragma unroll
        for (int i = 0; i < 16; ++i) {
          int key = keyb + i;
          vf[i] = qkv[(rowBase + kh * 256 + (key >> 2)) * NQKV + 1280 +
                      (size_t)(key & 3) * 64 + dcol];
        }
        oac[nt] = wmma32(pf, vf, oac[nt]);
      }
    }
    __syncthreads();
  }

  // ---- normalize + store (already in (g h d)-permuted columns) ----
#pragma unroll
  for (int nt = 0; nt < 4; ++nt)
#pragma unroll
    for (int r = 0; r < 8; ++r) {
      int l = l0 + r + 8 * (lane >> 4);
      int dd = nt * 16 + (lane & 15);
      o[(rowBase + l) * (size_t)DD + head * 64 + dd] =
          (f16)(oac[nt][r] / ls[r]);
    }
}

// ---------------------------------------------------------------------------
// ff = up * gelu_tanh(gate), stored f16 for down-proj WMMA
// ---------------------------------------------------------------------------
__global__ void gelu_mul_kernel(const float* __restrict__ up,
                                const float* __restrict__ gate,
                                f16* __restrict__ outh, size_t n) {
  size_t i = (size_t)blockIdx.x * blockDim.x + threadIdx.x;
  size_t stride = (size_t)gridDim.x * blockDim.x;
  for (; i < n; i += stride) {
    float x = gate[i];
    float t = tanhf(0.7978845608028654f * (x + 0.044715f * x * x * x));
    outh[i] = (f16)(up[i] * 0.5f * x * (1.0f + t));
  }
}

// ---------------------------------------------------------------------------
// LM head: only 4 rows (last position per batch) -> plain VALU dot products
// ---------------------------------------------------------------------------
__global__ __launch_bounds__(256) void lmhead_kernel(
    const f16* __restrict__ xh, const float* __restrict__ Wlm,
    const float* __restrict__ blm, float* __restrict__ out) {
  int v = blockIdx.x * 256 + threadIdx.x;
  int b = blockIdx.y;
  const f16* x = xh + ((size_t)b * TT + TT - 1) * DD;
  float acc = blm[v];
  for (int dd = 0; dd < DD; ++dd)
    acc += (float)x[dd] * Wlm[(size_t)dd * VV + v];
  out[(size_t)b * VV + v] = acc;
}

// ---------------------------------------------------------------------------
// Host orchestration
// ---------------------------------------------------------------------------
extern "C" void kernel_launch(void* const* d_in, const int* in_sizes, int n_in,
                              void* d_out, int out_size, void* d_ws,
                              size_t ws_size, hipStream_t stream) {
  (void)in_sizes; (void)n_in; (void)out_size; (void)ws_size;
  const int*   idx   = (const int*)d_in[0];
  const float* emb   = (const float*)d_in[1];
  const float* Wqkv  = (const float*)d_in[2];
  const float* Wout  = (const float*)d_in[3];
  const float* ln1g  = (const float*)d_in[4];
  const float* ln1b  = (const float*)d_in[5];
  const float* ln2g  = (const float*)d_in[6];
  const float* ln2b  = (const float*)d_in[7];
  const float* Wup   = (const float*)d_in[8];
  const float* bup   = (const float*)d_in[9];
  const float* Wgate = (const float*)d_in[10];
  const float* bgate = (const float*)d_in[11];
  const float* Wdown = (const float*)d_in[12];
  const float* bdown = (const float*)d_in[13];
  const float* flng  = (const float*)d_in[14];
  const float* flnb  = (const float*)d_in[15];
  const float* Wlm   = (const float*)d_in[16];
  const float* blm   = (const float*)d_in[17];
  float* out = (float*)d_out;

  char* ws = (char*)d_ws;
  size_t off = 0;
  auto alloc = [&](size_t bytes) -> char* {
    char* p = ws + off;
    off += (bytes + 255) & ~(size_t)255;
    return p;
  };

  const int M = BB * TT;  // 4096 token rows
  f16*   WqkvH  = (f16*)alloc((size_t)LL * DD * NQKV * 2);
  f16*   WoutH  = (f16*)alloc((size_t)LL * DD * DD * 2);
  f16*   WupH   = (f16*)alloc((size_t)LL * DD * FF * 2);
  f16*   WgateH = (f16*)alloc((size_t)LL * DD * FF * 2);
  f16*   WdownH = (f16*)alloc((size_t)LL * FF * DD * 2);
  float* h      = (float*)alloc((size_t)M * DD * 4);
  f16*   xh     = (f16*)alloc((size_t)M * DD * 2);
  f16*   qkvh   = (f16*)alloc((size_t)M * NQKV * 2);
  f16*   oh     = (f16*)alloc((size_t)M * DD * 2);
  float* upf    = (float*)alloc((size_t)M * FF * 4);
  float* gatef  = (float*)alloc((size_t)M * FF * 4);
  f16*   ffh    = (f16*)alloc((size_t)M * FF * 2);

  auto cvt = [&](const float* s, f16* d, size_t n) {
    int blocks = (int)((n + 2047) / 2048);
    cvt_kernel<<<blocks, 256, 0, stream>>>(s, d, n);
  };
  cvt(Wqkv,  WqkvH,  (size_t)LL * DD * NQKV);
  cvt(Wout,  WoutH,  (size_t)LL * DD * DD);
  cvt(Wup,   WupH,   (size_t)LL * DD * FF);
  cvt(Wgate, WgateH, (size_t)LL * DD * FF);
  cvt(Wdown, WdownH, (size_t)LL * FF * DD);

  embed_kernel<<<M, 256, 0, stream>>>(idx, emb, h);

  for (int l = 0; l < LL; ++l) {
    layernorm_kernel<<<M, 256, 0, stream>>>(h, ln1g + l * DD, ln1b + l * DD, xh);
    gemm_f16_kernel<<<dim3(NQKV / 128, M / 128), 256, 0, stream>>>(
        xh, WqkvH + (size_t)l * DD * NQKV, M, NQKV, DD,
        nullptr, nullptr, nullptr, qkvh);
    attention_kernel<<<dim3(TT / 128, BB * HH), 256, 0, stream>>>(qkvh, oh);
    gemm_f16_kernel<<<dim3(DD / 128, M / 128), 256, 0, stream>>>(
        oh, WoutH + (size_t)l * DD * DD, M, DD, DD,
        nullptr, h, h, nullptr);  // h += o @ Wout
    layernorm_kernel<<<M, 256, 0, stream>>>(h, ln2g + l * DD, ln2b + l * DD, xh);
    gemm_f16_kernel<<<dim3(FF / 128, M / 128), 256, 0, stream>>>(
        xh, WupH + (size_t)l * DD * FF, M, FF, DD,
        bup + (size_t)l * FF, nullptr, upf, nullptr);
    gemm_f16_kernel<<<dim3(FF / 128, M / 128), 256, 0, stream>>>(
        xh, WgateH + (size_t)l * DD * FF, M, FF, DD,
        bgate + (size_t)l * FF, nullptr, gatef, nullptr);
    gelu_mul_kernel<<<2048, 256, 0, stream>>>(upf, gatef, ffh, (size_t)M * FF);
    gemm_f16_kernel<<<dim3(DD / 128, M / 128), 256, 0, stream>>>(
        ffh, WdownH + (size_t)l * FF * DD, M, DD, FF,
        bdown + (size_t)l * DD, h, h, nullptr);  // h += ff @ Wdown + bd
  }

  layernorm_kernel<<<M, 256, 0, stream>>>(h, flng, flnb, xh);
  lmhead_kernel<<<dim3(VV / 256, BB), 256, 0, stream>>>(xh, Wlm, blm, out);
}
